// WFPNDeform_51951924412999
// MI455X (gfx1250) — compile-verified, matrix operand order checked by reading, not run on hardware
//
#include <hip/hip_runtime.h>
#include <hip/hip_bf16.h>

typedef __attribute__((ext_vector_type(16))) _Float16 v16h;
typedef __attribute__((ext_vector_type(8)))  _Float16 v8h;
typedef __attribute__((ext_vector_type(8)))  float    v8f;

union H16 { v16h v; v8h h2[2]; _Float16 e[16]; };

// A/B fragment for v_wmma_f32_16x16x32_f16: lanes 0-15 hold K {kk..kk+7, kk+16..kk+23},
// lanes 16-31 hold K {kk+8..kk+15, kk+24..kk+31} of row (m or n) = tile + (lane&15).
// Row-major K-contiguous storage -> two 16-byte global_load_b128.
__device__ inline v16h frag_load(const _Float16* __restrict__ row, int kk, int lane_hi) {
    H16 u;
    u.h2[0] = *(const v8h*)(row + kk + lane_hi * 8);
    u.h2[1] = *(const v8h*)(row + kk + 16 + lane_hi * 8);
    return u.v;
}

// ---------------------------------------------------------------------------
// Generic WMMA GEMM: C[M][N] = A[M][K](f16) * Bt[N][K](f16)^T (+bias)
// mode 0: f32 row-major store
// mode 1: relu -> f16 row-major store
// mode 2: relu -> f32 plane scatter: out[(row/S * N + col)*S + row%S]
// M must be a multiple of 128 (always 6400 here); K mult of 32; N arbitrary.
// N-edge handling: B row pointers are CLAMPED (not zeroed) — garbage columns
// n >= N are never stored, so the inner loop stays branch-free.
// ---------------------------------------------------------------------------
__global__ __launch_bounds__(256) void k_gemm_wmma(
    const _Float16* __restrict__ A, const _Float16* __restrict__ Bt,
    const float* __restrict__ bias, void* __restrict__ out,
    int M, int N, int K, int mode, int S)
{
    const int tid  = threadIdx.x;
    const int lane = tid & 31;
    const int wave = tid >> 5;
    const int wm = wave & 3, wn = wave >> 2;
    const int lane_lo = lane & 15, lane_hi = lane >> 4;

    const int m0 = blockIdx.y * 128 + wm * 16;   // second M tile at +64
    const int n0 = blockIdx.x * 64  + wn * 16;   // second N tile at +32

    v8f acc[2][2];
#pragma unroll
    for (int i = 0; i < 2; i++)
#pragma unroll
        for (int j = 0; j < 2; j++)
#pragma unroll
            for (int r = 0; r < 8; r++) acc[i][j][r] = 0.0f;

    const int nB0 = n0 + lane_lo, nB1 = n0 + 32 + lane_lo;
    const int nC0 = nB0 < N ? nB0 : N - 1;       // clamp: results masked at store
    const int nC1 = nB1 < N ? nB1 : N - 1;
    const _Float16* arow0 = A  + (size_t)(m0 + lane_lo) * K;
    const _Float16* arow1 = A  + (size_t)(m0 + 64 + lane_lo) * K;
    const _Float16* brow0 = Bt + (size_t)nC0 * K;
    const _Float16* brow1 = Bt + (size_t)nC1 * K;

    for (int kk = 0; kk < K; kk += 32) {
        v16h a0 = frag_load(arow0, kk, lane_hi);
        v16h a1 = frag_load(arow1, kk, lane_hi);
        v16h b0 = frag_load(brow0, kk, lane_hi);
        v16h b1 = frag_load(brow1, kk, lane_hi);
        int kp = (kk + 96 <= K) ? kk + 64 : kk;  // branch-free prefetch offset
        __builtin_prefetch(arow0 + kp, 0, 1);
        __builtin_prefetch(arow1 + kp, 0, 1);
        acc[0][0] = __builtin_amdgcn_wmma_f32_16x16x32_f16(false, a0, false, b0, (short)0, acc[0][0], false, false);
        acc[0][1] = __builtin_amdgcn_wmma_f32_16x16x32_f16(false, a0, false, b1, (short)0, acc[0][1], false, false);
        acc[1][0] = __builtin_amdgcn_wmma_f32_16x16x32_f16(false, a1, false, b0, (short)0, acc[1][0], false, false);
        acc[1][1] = __builtin_amdgcn_wmma_f32_16x16x32_f16(false, a1, false, b1, (short)0, acc[1][1], false, false);
    }

#pragma unroll
    for (int i = 0; i < 2; i++) {
        const int baseM = m0 + i * 64;
#pragma unroll
        for (int j = 0; j < 2; j++) {
            const int col = n0 + j * 32 + lane_lo;
            if (col >= N) continue;
            const float bv = bias ? bias[col] : 0.0f;
#pragma unroll
            for (int r = 0; r < 8; r++) {
                // C/D layout: VGPR r holds M = r (lanes 0-15) / r+8 (lanes 16-31)
                const int row = baseM + lane_hi * 8 + r;
                if (row >= M) continue;
                float v = acc[i][j][r] + bv;
                if (mode != 0) v = fmaxf(v, 0.0f);
                if (mode == 1) {
                    ((_Float16*)out)[(size_t)row * N + col] = (_Float16)v;
                } else if (mode == 2) {
                    const int bidx = row / S, sidx = row - bidx * S;
                    ((float*)out)[((size_t)(bidx * N + col)) * S + sidx] = v;
                } else {
                    ((float*)out)[(size_t)row * N + col] = v;
                }
            }
        }
    }
}

// ---------------------------------------------------------------------------
// ori_fe = mean(maxpool4(x0), maxpool2(x1), x2, nn(x3), nn(x4)) at 40x40
// ---------------------------------------------------------------------------
__global__ void k_build_ori(const float* __restrict__ x0, const float* __restrict__ x1,
                            const float* __restrict__ x2, const float* __restrict__ x3,
                            const float* __restrict__ x4, float* __restrict__ ori)
{
    int t = blockIdx.x * 256 + threadIdx.x;
    if (t >= 4 * 256 * 1600) return;
    int bc = t / 1600;
    int s  = t - bc * 1600;
    int y = s / 40, x = s - y * 40;

    const float* p0 = x0 + (size_t)bc * 25600;
    float m0 = -3.402823466e38f;
#pragma unroll
    for (int dy = 0; dy < 4; dy++)
#pragma unroll
        for (int dx = 0; dx < 4; dx++)
            m0 = fmaxf(m0, p0[(4 * y + dy) * 160 + 4 * x + dx]);

    const float* p1 = x1 + (size_t)bc * 6400;
    float m1 = -3.402823466e38f;
#pragma unroll
    for (int dy = 0; dy < 2; dy++)
#pragma unroll
        for (int dx = 0; dx < 2; dx++)
            m1 = fmaxf(m1, p1[(2 * y + dy) * 80 + 2 * x + dx]);

    float f2 = x2[(size_t)bc * 1600 + s];
    float f3 = x3[(size_t)bc * 400 + (y >> 1) * 20 + (x >> 1)];
    float f4 = x4[(size_t)bc * 100 + (y >> 2) * 10 + (x >> 2)];
    ori[t] = (m0 + m1 + f2 + f3 + f4) * 0.2f;
}

__global__ void k_f32_to_f16(const float* __restrict__ in, _Float16* __restrict__ out, int n)
{
    int t = blockIdx.x * 256 + threadIdx.x;
    if (t < n) out[t] = (_Float16)in[t];
}

// im2col for 3x3 pad-1 conv at 40x40: A[m][c*9+k]
__global__ void k_im2col(const float* __restrict__ ori, _Float16* __restrict__ A)
{
    int t = blockIdx.x * 256 + threadIdx.x;
    if (t >= 6400 * 2304) return;
    int m = t / 2304, e = t - m * 2304;
    int c = e / 9,    k = e - c * 9;
    int b = m / 1600; int s = m - b * 1600;
    int y = s / 40, x = s - y * 40;
    int yy = y + k / 3 - 1;
    int xx = x + k % 3 - 1;
    float v = 0.0f;
    if ((unsigned)yy < 40u && (unsigned)xx < 40u)
        v = ori[((size_t)(b * 256 + c)) * 1600 + yy * 40 + xx];
    A[(size_t)m * 2304 + e] = (_Float16)v;
}

// Deformable bilinear sampling: sample point p for pixel (y,x) is (y+pts_y, x+pts_x)
// (the base kernel offsets cancel against dcn_offset). One block per pixel m.
__global__ __launch_bounds__(256) void k_deform(const float* __restrict__ ori,
                                                const float* __restrict__ pts,
                                                _Float16* __restrict__ A)
{
    int m = blockIdx.x;
    int b = m / 1600; int s = m - b * 1600;
    int y = s / 40, x = s - y * 40;

    __shared__ int   siy[9], six[9];
    __shared__ float swy[9], swx[9];
    if (threadIdx.x < 9) {
        int p = threadIdx.x;
        float py = (float)y + pts[(size_t)m * 18 + 2 * p];
        float px = (float)x + pts[(size_t)m * 18 + 2 * p + 1];
        float fy = floorf(py), fx = floorf(px);
        siy[p] = (int)fy; six[p] = (int)fx;
        swy[p] = py - fy; swx[p] = px - fx;
    }
    __syncthreads();

    const float* pb = ori + (size_t)b * 256 * 1600;
    for (int it = 0; it < 9; it++) {
        int e = threadIdx.x + it * 256;      // 0..2303
        int c = e / 9, p = e - c * 9;
        const float* pl = pb + (size_t)c * 1600;
        int iy0 = siy[p], ix0 = six[p];
        float wy1 = swy[p], wx1 = swx[p];
        float wy0 = 1.0f - wy1, wx0 = 1.0f - wx1;
        float v = 0.0f;
        if ((unsigned)iy0 < 40u) {
            if ((unsigned)ix0 < 40u)       v += pl[iy0 * 40 + ix0]           * wy0 * wx0;
            if ((unsigned)(ix0 + 1) < 40u) v += pl[iy0 * 40 + ix0 + 1]       * wy0 * wx1;
        }
        if ((unsigned)(iy0 + 1) < 40u) {
            if ((unsigned)ix0 < 40u)       v += pl[(iy0 + 1) * 40 + ix0]     * wy1 * wx0;
            if ((unsigned)(ix0 + 1) < 40u) v += pl[(iy0 + 1) * 40 + ix0 + 1] * wy1 * wx1;
        }
        A[(size_t)m * 2304 + e] = (_Float16)v;
    }
}

// Per-level fusion: out = xi + nn_resize(bsf) * (tanh(relu(conv1)) + tanh(relu(conv2)))
// 1-output-channel 3x3 convs over 256 input channels (VALU; N=1 makes WMMA 15/16 idle).
__global__ void k_fuse(const float* __restrict__ xi,
                       const float* __restrict__ w1, const float* __restrict__ b1p,
                       const float* __restrict__ w2, const float* __restrict__ b2p,
                       const float* __restrict__ bsf, float* __restrict__ out,
                       int H, int W)
{
    int t = blockIdx.x * blockDim.x + threadIdx.x;
    int HW = H * W;
    if (t >= 4 * HW) return;
    int b = t / HW; int s = t - b * HW;
    int y = s / W, x = s - y * W;

    const float* xb = xi + (size_t)b * 256 * HW;
    float a1 = b1p[0], a2 = b2p[0];
    for (int c = 0; c < 256; c++) {
        const float* pl  = xb + (size_t)c * HW;
        const float* wc1 = w1 + c * 9;
        const float* wc2 = w2 + c * 9;
#pragma unroll
        for (int ky = 0; ky < 3; ky++) {
            int yy = y + ky - 1;
            if ((unsigned)yy >= (unsigned)H) continue;
#pragma unroll
            for (int kx = 0; kx < 3; kx++) {
                int xx = x + kx - 1;
                if ((unsigned)xx >= (unsigned)W) continue;
                float v = pl[yy * W + xx];
                a1 += v * wc1[ky * 3 + kx];
                a2 += v * wc2[ky * 3 + kx];
            }
        }
    }
    float g = tanhf(fmaxf(a1, 0.0f)) + tanhf(fmaxf(a2, 0.0f));
    int ys = (y * 40) / H, xs = (x * 40) / W;
    const float* bb = bsf + (size_t)b * 256 * 1600;
    for (int c = 0; c < 256; c++) {
        out[((size_t)(b * 256 + c)) * HW + s] =
            xb[(size_t)c * HW + s] + bb[c * 1600 + ys * 40 + xs] * g;
    }
}

extern "C" void kernel_launch(void* const* d_in, const int* in_sizes, int n_in,
                              void* d_out, int out_size, void* d_ws, size_t ws_size,
                              hipStream_t stream)
{
    (void)in_sizes; (void)n_in; (void)out_size; (void)ws_size;
    const float* x0         = (const float*)d_in[0];
    const float* x1         = (const float*)d_in[1];
    const float* x2         = (const float*)d_in[2];
    const float* x3         = (const float*)d_in[3];
    const float* x4         = (const float*)d_in[4];
    const float* w_red1     = (const float*)d_in[5];
    const float* b_red1     = (const float*)d_in[6];
    const float* w_red2     = (const float*)d_in[7];
    const float* b_red2     = (const float*)d_in[8];
    const float* w_pts_conv = (const float*)d_in[9];
    const float* b_pts_conv = (const float*)d_in[10];
    const float* w_pts_out  = (const float*)d_in[11];
    const float* b_pts_out  = (const float*)d_in[12];
    const float* w_refine   = (const float*)d_in[13];

    char* ws = (char*)d_ws;
    auto align256 = [](size_t v) { return (v + 255) & ~(size_t)255; };
    size_t off = 0;
    float*    ori    = (float*)(ws + off);    off = align256(off + (size_t)4 * 256 * 1600 * 4);
    _Float16* Abuf   = (_Float16*)(ws + off); off = align256(off + (size_t)6400 * 2304 * 2);
    _Float16* WtPts  = (_Float16*)(ws + off); off = align256(off + (size_t)256 * 2304 * 2);
    _Float16* WtRef  = (_Float16*)(ws + off); off = align256(off + (size_t)256 * 2304 * 2);
    _Float16* WtPO   = (_Float16*)(ws + off); off = align256(off + (size_t)18 * 256 * 2);
    _Float16* hidden = (_Float16*)(ws + off); off = align256(off + (size_t)6400 * 256 * 2);
    float*    pts    = (float*)(ws + off);    off = align256(off + (size_t)6400 * 18 * 4);
    float*    bsf    = (float*)(ws + off);    off = align256(off + (size_t)4 * 256 * 1600 * 4);

    // 1. fused pyramid pooling -> ori_fe (f32 planes)
    k_build_ori<<<6400, 256, 0, stream>>>(x0, x1, x2, x3, x4, ori);

    // 2. weight packs: (O,C,3,3) row-major is already [O][K=C*9] -> convert to f16
    k_f32_to_f16<<<(589824 + 255) / 256, 256, 0, stream>>>(w_pts_conv, WtPts, 589824);
    k_f32_to_f16<<<(589824 + 255) / 256, 256, 0, stream>>>(w_refine,   WtRef, 589824);
    k_f32_to_f16<<<(4608 + 255) / 256,   256, 0, stream>>>(w_pts_out,  WtPO,  4608);

    // 3. im2col of ori_fe
    k_im2col<<<(6400 * 2304) / 256, 256, 0, stream>>>(ori, Abuf);

    // 4. pts_conv GEMM (M=6400,N=256,K=2304) + bias + relu -> f16 hidden
    dim3 gBig(256 / 64, 6400 / 128);
    k_gemm_wmma<<<gBig, 256, 0, stream>>>(Abuf, WtPts, b_pts_conv, hidden, 6400, 256, 2304, 1, 1);

    // 5. pts_out 1x1 GEMM (M=6400,N=18,K=256) + bias -> f32 pts[m][18]
    dim3 gSmall(1, 6400 / 128);
    k_gemm_wmma<<<gSmall, 256, 0, stream>>>(hidden, WtPO, b_pts_out, pts, 6400, 18, 256, 0, 1);

    // 6. deformable bilinear sampling -> Abuf reused as [6400][2304]
    k_deform<<<6400, 256, 0, stream>>>(ori, pts, Abuf);

    // 7. refine GEMM + relu -> bsf f32 planes [b][256][40][40]
    k_gemm_wmma<<<gBig, 256, 0, stream>>>(Abuf, WtRef, nullptr, bsf, 6400, 256, 2304, 2, 1600);

    // 8. per-level fusion into d_out (outputs concatenated in level order)
    const float* xs[5] = {x0, x1, x2, x3, x4};
    const int    Hs[5] = {160, 80, 40, 20, 10};
    float* outp = (float*)d_out;
    size_t oo = 0;
    for (int i = 0; i < 5; i++) {
        int H = Hs[i], W = Hs[i];
        int tot = 4 * H * W;
        k_fuse<<<(tot + 255) / 256, 256, 0, stream>>>(
            xs[i], w_red1 + (size_t)i * 2304, b_red1 + i,
            w_red2 + (size_t)i * 2304, b_red2 + i,
            bsf, outp + oo, H, W);
        oo += (size_t)4 * 256 * H * W;
    }
}
